// mRNN_28106265985346
// MI455X (gfx1250) — compile-verified, matrix operand order checked by reading, not run on hardware
//
#include <hip/hip_runtime.h>
#include <hip/hip_bf16.h>

typedef __attribute__((ext_vector_type(16))) __bf16 v16bf;
typedef __attribute__((ext_vector_type(8)))  float  v8f;
typedef __attribute__((ext_vector_type(4)))  unsigned int v4u;
typedef __attribute__((ext_vector_type(8)))  int    v8i;
typedef __attribute__((ext_vector_type(4)))  int    v4i;

constexpr int Bb = 64, Tt = 300, Nn = 1024, Ii = 128;
constexpr float ALPHA       = 0.1f;
constexpr float NOISE_SCALE = 0.44721359549995794f;   // sqrt(2*ALPHA)
constexpr float CREC        = NOISE_SCALE * 0.01f;    // NOISE_SCALE*SIGMA_REC
constexpr float CINP        = NOISE_SCALE * 0.01f;    // NOISE_SCALE*SIGMA_INP

constexpr int NUM_WG = 16;        // each owns 64 columns of N

// LDS h staging: row-major [64][1024] bf16 with TDM pad: +16B per 1KB chunk
// => row stride 2080 B (16B aligned, 2-way max bank conflict for A fetches)
constexpr int HROW = 2080;

// workspace layout (bytes)
constexpr size_t OFF_CTR = 0;
constexpr size_t OFF_WR  = 1024;                              // bf16 [N][N]   2 MB
constexpr size_t OFF_WI  = OFF_WR + (size_t)Nn * Nn * 2;      // bf16 [N][I]   256 KB
constexpr size_t OFF_H0  = OFF_WI + (size_t)Nn * Ii * 2;      // bf16 [B][N]   128 KB
constexpr size_t OFF_H1  = OFF_H0 + (size_t)Bb * Nn * 2;      // bf16 [B][N]   128 KB

// ---------------------------------------------------------------------------
// Prep: effective Dale's-law weights -> bf16, h0 -> bf16, zero the sync counter
// ---------------------------------------------------------------------------
__global__ void mrnn_prep(const float* __restrict__ W_rec, const float* __restrict__ W_inp,
                          const float* __restrict__ sign_rec, const float* __restrict__ mask_rec,
                          const float* __restrict__ sign_inp, const float* __restrict__ mask_inp,
                          const float* __restrict__ h0,
                          __bf16* __restrict__ Wr_bf, __bf16* __restrict__ Wi_bf,
                          __bf16* __restrict__ hbf0, unsigned* __restrict__ counter) {
    int idx = blockIdx.x * 256 + threadIdx.x;
    if (idx == 0) *counter = 0u;
    if (idx < Nn * Nn) {
        float w = W_rec[idx];
        w = fminf(fmaxf(w, 0.0f), 10.0f) * sign_rec[idx] * mask_rec[idx];
        Wr_bf[idx] = (__bf16)w;
    } else if (idx < Nn * Nn + Nn * Ii) {
        int j = idx - Nn * Nn;
        float w = W_inp[j];
        w = fminf(fmaxf(w, 0.0f), 10.0f) * sign_inp[j] * mask_inp[j];
        Wi_bf[j] = (__bf16)w;
    } else if (idx < Nn * Nn + Nn * Ii + Bb * Nn) {
        int j = idx - Nn * Nn - Nn * Ii;
        hbf0[j] = (__bf16)h0[j];
    }
}

// ---------------------------------------------------------------------------
// Device-wide barrier (16 resident workgroups, atomic counter in workspace)
// ---------------------------------------------------------------------------
__device__ __forceinline__ void grid_sync(unsigned* counter, unsigned target) {
    __threadfence();
    __syncthreads();
    if (threadIdx.x == 0) {
        __hip_atomic_fetch_add(counter, 1u, __ATOMIC_RELEASE, __HIP_MEMORY_SCOPE_AGENT);
        while (__hip_atomic_load(counter, __ATOMIC_ACQUIRE, __HIP_MEMORY_SCOPE_AGENT) < target) {
            __builtin_amdgcn_s_sleep(2);
        }
    }
    __syncthreads();
}

// ---------------------------------------------------------------------------
// Persistent recurrent kernel: 16 WGs x 256 threads (8 waves).
// Per step: TDM DMAs h (64x1024 bf16) into padded LDS while the waves build
// u and run the K=128 input GEMM; then s_wait_tensorcnt and run the K=1024
// recurrent GEMM with A from LDS, B (Wr) from L2. h stays fp32 in VGPRs.
// ---------------------------------------------------------------------------
__launch_bounds__(256, 1)
__global__ void mrnn_scan(const float* __restrict__ x, const float* __restrict__ h0,
                          const float* __restrict__ base_firing,
                          const float* __restrict__ noise_rec,
                          const float* __restrict__ noise_inp,
                          const __bf16* __restrict__ Wr_bf, const __bf16* __restrict__ Wi_bf,
                          __bf16* __restrict__ hbf0, __bf16* __restrict__ hbf1,
                          float* __restrict__ out, unsigned* __restrict__ counter) {
    __shared__ __attribute__((aligned(1024))) char   h_lds[Bb * HROW];   // 130 KB padded h
    __shared__ __attribute__((aligned(128)))  __bf16 u_lds[Bb * Ii];     // 16 KB noisy input

    const int tid   = threadIdx.x;
    const int wave  = tid >> 5;
    const int lane  = tid & 31;
    const int l16   = lane & 15;
    const int lhalf = lane >> 4;

    const int m0  = (wave & 3) * 16;                 // batch-row tile base
    const int nt0 = blockIdx.x * 64 + (wave >> 2) * 32;
    const int nt1 = nt0 + 16;

    // C/D tile element coordinates for this lane (ISA 16x16 f32 layout)
    const int brow  = m0 + (lhalf << 3);             // + r gives batch index
    const int ncol0 = nt0 + l16;
    const int ncol1 = nt1 + l16;

    // fp32 hidden state resident in registers for the whole scan
    v8f h0r, h1r;
#pragma unroll
    for (int r = 0; r < 8; ++r) {
        h0r[r] = h0[(brow + r) * Nn + ncol0];
        h1r[r] = h0[(brow + r) * Nn + ncol1];
    }
    const float bf0 = base_firing[ncol0];
    const float bf1 = base_firing[ncol1];

    // operand base addressing
    const int arow  = (m0 + l16) * HROW;             // padded LDS row base for A
    const int boff0 = (nt0 + l16) * Nn + lhalf * 16;
    const int boff1 = (nt1 + l16) * Nn + lhalf * 16;
    const __bf16* Wi0  = Wi_bf + (nt0 + l16) * Ii + lhalf * 16;
    const __bf16* Wi1  = Wi_bf + (nt1 + l16) * Ii + lhalf * 16;
    const __bf16* Ulds = u_lds + (m0 + l16) * Ii + lhalf * 16;

    const unsigned lds_h_base = (unsigned)(unsigned long long)(const void*)&h_lds[0];

    for (int t = 0; t < Tt; ++t) {
        const __bf16* hcur  = (t & 1) ? hbf1 : hbf0;
        __bf16*       hnext = (t & 1) ? hbf0 : hbf1;

        // --- wave 0: kick TDM DMA of h[cur] (64x1024 bf16) -> padded LDS ---
        if (tid < 32) {
            unsigned long long ga = (unsigned long long)(const void*)hcur;
            // D# group0: count=1, lds_addr, global_addr[56:0], type=2
            v4u g0 = { 1u, lds_h_base,
                       (unsigned)(ga & 0xffffffffu),
                       (unsigned)((ga >> 32) & 0x1ffffffu) | (2u << 30) };
            // D# group1: data_size=2B, pad_enable, interval=256dw, amount=4dw,
            // tensor_dim0=1024, tensor_dim1=64, tile_dim0=1024, tile_dim1=64,
            // tensor_dim0_stride=1024
            v8i g1 = { (int)0x07D10000,   // wg_mask=0|data_size=1|pad|intv=7|amt=3
                       (int)0x04000000,   // tensor_dim0[15:0]=1024 << 16
                       (int)0x00400000,   // tensor_dim1[15:0]=64  << 16
                       (int)0x04000000,   // tile_dim0=1024 << 16
                       64,                // tile_dim1=64, tile_dim2=0
                       1024,              // tensor_dim0_stride[31:0]
                       0, 0 };
            v4i g2 = { 0, 0, 0, 0 };
            v4i g3 = { 0, 0, 0, 0 };
            v8i g4 = { 0, 0, 0, 0, 0, 0, 0, 0 };   // extra group (2D tensor: unused)
            __builtin_amdgcn_tensor_load_to_lds(g0, g1, g2, g3, g4, 0);
        }

        // --- all waves: stage noisy input u = x_t + c*ni into LDS ----------
#pragma unroll 4
        for (int j = 0; j < (Bb * Ii) / 256; ++j) {
            int idx = tid + j * 256;          // idx == b*128 + i
            int b = idx >> 7, i = idx & 127;
            float u = x[b * (Tt * Ii) + t * Ii + i]
                    + CINP * noise_inp[t * (Bb * Ii) + b * Ii + i];
            u_lds[idx] = (__bf16)u;
        }
        __syncthreads();

        v8f acc0 = {}, acc1 = {};

        // --- input GEMM first (K=128, A from LDS u) — hides TDM latency ----
#pragma unroll
        for (int k = 0; k < Ii; k += 32) {
            v16bf a  = *(const v16bf*)(Ulds + k);
            v16bf b0 = *(const v16bf*)(Wi0 + k);
            v16bf b1 = *(const v16bf*)(Wi1 + k);
            acc0 = __builtin_amdgcn_wmma_f32_16x16x32_bf16(false, a, false, b0,
                                                           (short)0, acc0, false, false);
            acc1 = __builtin_amdgcn_wmma_f32_16x16x32_bf16(false, a, false, b1,
                                                           (short)0, acc1, false, false);
        }

        // --- wait for TDM, then recurrent GEMM (K=1024, A from padded LDS) -
        if (tid < 32) __builtin_amdgcn_s_wait_tensorcnt(0);
        __syncthreads();

        const __bf16* B0 = Wr_bf + boff0;
        const __bf16* B1 = Wr_bf + boff1;
#pragma unroll 4
        for (int k = 0; k < Nn; k += 32) {
            const int kel = k + lhalf * 16;                    // element index in row
            const int abyte = arow + 2 * kel + ((kel >> 9) << 4);  // +16B pad per 1KB
            v16bf a  = *(const v16bf*)(h_lds + abyte);
            v16bf b0 = *(const v16bf*)(B0 + k);
            v16bf b1 = *(const v16bf*)(B1 + k);
            __builtin_prefetch(B0 + k + 256, 0, 3);
            __builtin_prefetch(B1 + k + 256, 0, 3);
            acc0 = __builtin_amdgcn_wmma_f32_16x16x32_bf16(false, a, false, b0,
                                                           (short)0, acc0, false, false);
            acc1 = __builtin_amdgcn_wmma_f32_16x16x32_bf16(false, a, false, b1,
                                                           (short)0, acc1, false, false);
        }

        // --- pointwise update, stores (fp32 out, bf16 next-h) --------------
        const int tbase = t * (Bb * Nn);
#pragma unroll
        for (int r = 0; r < 8; ++r) {
            const int b = brow + r;
            float pre0 = acc0[r] + bf0;
            float pre1 = acc1[r] + bf1;
            float nr0 = noise_rec[tbase + b * Nn + ncol0];
            float nr1 = noise_rec[tbase + b * Nn + ncol1];
            float hn0 = (1.0f - ALPHA) * h0r[r] + ALPHA * (fmaxf(pre0, 0.0f) + CREC * nr0);
            float hn1 = (1.0f - ALPHA) * h1r[r] + ALPHA * (fmaxf(pre1, 0.0f) + CREC * nr1);
            h0r[r] = hn0;
            h1r[r] = hn1;
            out[b * (Tt * Nn) + t * Nn + ncol0] = hn0;
            out[b * (Tt * Nn) + t * Nn + ncol1] = hn1;
            hnext[b * Nn + ncol0] = (__bf16)hn0;
            hnext[b * Nn + ncol1] = (__bf16)hn1;
        }

        // --- one device-wide barrier per step ------------------------------
        grid_sync(counter, (unsigned)(NUM_WG * (t + 1)));
    }
}

// ---------------------------------------------------------------------------
extern "C" void kernel_launch(void* const* d_in, const int* in_sizes, int n_in,
                              void* d_out, int out_size, void* d_ws, size_t ws_size,
                              hipStream_t stream) {
    const float* x           = (const float*)d_in[0];
    const float* h0          = (const float*)d_in[1];
    const float* W_rec       = (const float*)d_in[2];
    const float* W_inp       = (const float*)d_in[3];
    const float* sign_rec    = (const float*)d_in[4];
    const float* mask_rec    = (const float*)d_in[5];
    const float* sign_inp    = (const float*)d_in[6];
    const float* mask_inp    = (const float*)d_in[7];
    const float* base_firing = (const float*)d_in[8];
    const float* noise_rec   = (const float*)d_in[9];
    const float* noise_inp   = (const float*)d_in[10];
    float* out = (float*)d_out;

    char* ws = (char*)d_ws;
    unsigned* counter = (unsigned*)(ws + OFF_CTR);
    __bf16*   Wr_bf   = (__bf16*)(ws + OFF_WR);
    __bf16*   Wi_bf   = (__bf16*)(ws + OFF_WI);
    __bf16*   hbf0    = (__bf16*)(ws + OFF_H0);
    __bf16*   hbf1    = (__bf16*)(ws + OFF_H1);

    const int total = Nn * Nn + Nn * Ii + Bb * Nn;
    mrnn_prep<<<(total + 255) / 256, 256, 0, stream>>>(
        W_rec, W_inp, sign_rec, mask_rec, sign_inp, mask_inp, h0,
        Wr_bf, Wi_bf, hbf0, counter);

    mrnn_scan<<<NUM_WG, 256, 0, stream>>>(
        x, h0, base_firing, noise_rec, noise_inp,
        Wr_bf, Wi_bf, hbf0, hbf1, out, counter);
}
